// MelSpectrogram_75118978007174
// MI455X (gfx1250) — compile-verified, matrix operand order checked by reading, not run on hardware
//
#include <hip/hip_runtime.h>
#include <hip/hip_bf16.h>

// ---------------- problem constants ----------------
#define BATCH     32
#define AUDIO_LEN 220500
#define NFFT      2048
#define HOP       512
#define FREQ      1025      // NFFT/2 + 1
#define NMELS     128
#define TFRAMES   431
#define PADL      (NFFT / 2)                 // 1024
#define LPAD      (AUDIO_LEN + NFFT)         // 222548
#define XSTRIDE   222560                     // LPAD rounded up, 16B-aligned rows (bf16)
#define FPADW     1056                       // weight rows padded: 66 freq tiles
#define SPADF     1056                       // spec width = FPADW (multiple of 32)
#define KTOT      NFFT
#define MWG       32                         // frames per workgroup (2 sub-tiles of 16)
#define TILES_PER_B 14                       // ceil(431/32)
#define WINHALFS  ((MWG - 1) * HOP + NFFT)   // 17920 samples staged per WG

typedef unsigned int u32x4 __attribute__((ext_vector_type(4), aligned(16)));
typedef float        v8f   __attribute__((ext_vector_type(8)));
typedef __bf16       v16bf __attribute__((ext_vector_type(16)));

union Frag {
    u32x4 u[2];
    v16bf v;
};

// ---------------- prep kernels (bf16 staging in workspace) ----------------

__global__ __launch_bounds__(256) void k_pad_bf16(const float* __restrict__ x,
                                                  __bf16* __restrict__ xpad) {
    int idx = blockIdx.x * 256 + threadIdx.x;
    const int total = BATCH * LPAD;
    if (idx >= total) return;
    int b = idx / LPAD;
    int i = idx - b * LPAD;
    int j = i - PADL;
    if (j < 0) j = -j;
    else if (j >= AUDIO_LEN) j = 2 * AUDIO_LEN - 2 - j;
    xpad[(size_t)b * XSTRIDE + i] = (__bf16)x[(size_t)b * AUDIO_LEN + j];
}

// Wcos_t / Wsin_t : [FPADW][KTOT] bf16, rows >= FREQ zeroed.
__global__ __launch_bounds__(256) void k_weights_bf16(const float* __restrict__ wsin,
                                                      const float* __restrict__ wcos,
                                                      __bf16* __restrict__ Wc,
                                                      __bf16* __restrict__ Ws) {
    int idx = blockIdx.x * 256 + threadIdx.x;
    if (idx >= FPADW * KTOT) return;
    int f = idx / KTOT;
    float c = 0.f, s = 0.f;
    if (f < FREQ) { c = wcos[idx]; s = wsin[idx]; }
    Wc[idx] = (__bf16)c;
    Ws[idx] = (__bf16)s;
}

// mel basis -> bf16 [NMELS][SPADF], cols >= FREQ zeroed
__global__ __launch_bounds__(256) void k_mel_bf16(const float* __restrict__ melb_f,
                                                  __bf16* __restrict__ melb) {
    int idx = blockIdx.x * 256 + threadIdx.x;
    if (idx >= NMELS * SPADF) return;
    int m = idx / SPADF;
    int f = idx - m * SPADF;
    float v = (f < FREQ) ? melb_f[(size_t)m * FREQ + f] : 0.f;
    melb[idx] = (__bf16)v;
}

// ---------------- fused STFT-power + mel kernel ----------------
// One workgroup (8 wave32) per 32-frame tile of one batch row.
// A-tile stored as the overlapping audio window: A[m][k] = win[m*HOP + k].
// Each wave processes 2 freq tiles x 2 M sub-tiles per pass (8 accumulators),
// so each B fragment feeds 2 WMMAs and each A fragment feeds 4.
__global__ __launch_bounds__(256) void k_melspec(const __bf16* __restrict__ xpad,
                                                 const __bf16* __restrict__ Wc,
                                                 const __bf16* __restrict__ Ws,
                                                 const __bf16* __restrict__ melb,
                                                 float* __restrict__ out) {
    __shared__ __bf16 sWin[WINHALFS];       // 35.8 KB audio window
    __shared__ __bf16 sSpec[MWG * SPADF];   // 66 KB bf16 power spectrum tile

    const int tid   = threadIdx.x;
    const int wave  = tid >> 5;
    const int lane  = tid & 31;
    const int nrow  = lane & 15;            // A: M row / B: N column within tile
    const int hi    = lane >> 4;
    const int koff  = hi ? 8 : 0;           // 16-bit fragment K sub-offset
    const int b     = blockIdx.x / TILES_PER_B;
    const int t0    = (blockIdx.x % TILES_PER_B) * MWG;

    // ---- stage overlapping audio window into LDS ----
    {
        const int base = t0 * HOP;          // halves into padded row
        const __bf16* row = xpad + (size_t)b * XSTRIDE;
        for (int q = tid; q < WINHALFS / 8; q += 256) {
            int s = base + q * 8;
            if (s > LPAD - 8) s = (LPAD - 8) & ~7;   // keep 16B-aligned, in-row
            ((u32x4*)sWin)[q] = *(const u32x4*)(row + s);
        }
    }
    __syncthreads();

    // ---- STFT GEMM: 33 freq-tile pairs round-robin over 8 waves ----
    const __bf16* arow0 = sWin + nrow * HOP;
    const __bf16* arow1 = sWin + (nrow + 16) * HOP;
    for (int fp = wave; fp < SPADF / 32; fp += 8) {
        const int ft0 = fp * 2;
        // second tile's rows are a constant +16*KTOT from the first
        const __bf16* bC = Wc + (size_t)(ft0 * 16 + nrow) * KTOT;
        const __bf16* bS = Ws + (size_t)(ft0 * 16 + nrow) * KTOT;
        __builtin_prefetch(bC, 0, 1);
        __builtin_prefetch(bS, 0, 1);
        v8f aR00 = {}, aI00 = {}, aR10 = {}, aI10 = {};   // sub-tile 0/1 x freq tile 0
        v8f aR01 = {}, aI01 = {}, aR11 = {}, aI11 = {};   // sub-tile 0/1 x freq tile 1
        for (int kc = 0; kc < KTOT; kc += 32) {
            Frag a0, a1, bc0, bs0, bc1, bs1;
            a0.u[0]  = *(const u32x4*)(arow0 + kc + koff);
            a0.u[1]  = *(const u32x4*)(arow0 + kc + 16 + koff);
            a1.u[0]  = *(const u32x4*)(arow1 + kc + koff);
            a1.u[1]  = *(const u32x4*)(arow1 + kc + 16 + koff);
            bc0.u[0] = *(const u32x4*)(bC + kc + koff);
            bc0.u[1] = *(const u32x4*)(bC + kc + 16 + koff);
            bs0.u[0] = *(const u32x4*)(bS + kc + koff);
            bs0.u[1] = *(const u32x4*)(bS + kc + 16 + koff);
            bc1.u[0] = *(const u32x4*)(bC + 16 * KTOT + kc + koff);
            bc1.u[1] = *(const u32x4*)(bC + 16 * KTOT + kc + 16 + koff);
            bs1.u[0] = *(const u32x4*)(bS + 16 * KTOT + kc + koff);
            bs1.u[1] = *(const u32x4*)(bS + 16 * KTOT + kc + 16 + koff);
            aR00 = __builtin_amdgcn_wmma_f32_16x16x32_bf16(false, a0.v, false, bc0.v,
                                                           (short)0, aR00, false, false);
            aI00 = __builtin_amdgcn_wmma_f32_16x16x32_bf16(false, a0.v, false, bs0.v,
                                                           (short)0, aI00, false, false);
            aR10 = __builtin_amdgcn_wmma_f32_16x16x32_bf16(false, a1.v, false, bc0.v,
                                                           (short)0, aR10, false, false);
            aI10 = __builtin_amdgcn_wmma_f32_16x16x32_bf16(false, a1.v, false, bs0.v,
                                                           (short)0, aI10, false, false);
            aR01 = __builtin_amdgcn_wmma_f32_16x16x32_bf16(false, a0.v, false, bc1.v,
                                                           (short)0, aR01, false, false);
            aI01 = __builtin_amdgcn_wmma_f32_16x16x32_bf16(false, a0.v, false, bs1.v,
                                                           (short)0, aI01, false, false);
            aR11 = __builtin_amdgcn_wmma_f32_16x16x32_bf16(false, a1.v, false, bc1.v,
                                                           (short)0, aR11, false, false);
            aI11 = __builtin_amdgcn_wmma_f32_16x16x32_bf16(false, a1.v, false, bs1.v,
                                                           (short)0, aI11, false, false);
        }
        const int f0 = ft0 * 16 + nrow;
#pragma unroll
        for (int r = 0; r < 8; ++r) {
            float re, im;
            re = aR00[r]; im = aI00[r];
            sSpec[(r + hi * 8) * SPADF + f0]           = (__bf16)(re * re + im * im);
            re = aR10[r]; im = aI10[r];
            sSpec[(16 + r + hi * 8) * SPADF + f0]      = (__bf16)(re * re + im * im);
            re = aR01[r]; im = aI01[r];
            sSpec[(r + hi * 8) * SPADF + f0 + 16]      = (__bf16)(re * re + im * im);
            re = aR11[r]; im = aI11[r];
            sSpec[(16 + r + hi * 8) * SPADF + f0 + 16] = (__bf16)(re * re + im * im);
        }
    }
    __syncthreads();

    // ---- mel GEMM: wave w handles mel rows [w*16, w*16+16) vs both sub-tiles ----
    {
        const __bf16* amel = melb + (size_t)(wave * 16 + nrow) * SPADF;
#pragma unroll
        for (int ms = 0; ms < 2; ++ms) {
            const __bf16* brow = sSpec + (ms * 16 + nrow) * SPADF;
            v8f acc = {};
            for (int kc = 0; kc < SPADF; kc += 32) {
                Frag a, bb;
                a.u[0]  = *(const u32x4*)(amel + kc + koff);
                a.u[1]  = *(const u32x4*)(amel + kc + 16 + koff);
                bb.u[0] = *(const u32x4*)(brow + kc + koff);
                bb.u[1] = *(const u32x4*)(brow + kc + 16 + koff);
                acc = __builtin_amdgcn_wmma_f32_16x16x32_bf16(false, a.v, false, bb.v,
                                                              (short)0, acc, false, false);
            }
            const int t = t0 + ms * 16 + nrow;
            if (t < TFRAMES) {
#pragma unroll
                for (int r = 0; r < 8; ++r) {
                    int mel = wave * 16 + r + hi * 8;
                    out[((size_t)b * NMELS + mel) * TFRAMES + t] = acc[r];
                }
            }
        }
    }
}

// ---------------- launch ----------------
extern "C" void kernel_launch(void* const* d_in, const int* in_sizes, int n_in,
                              void* d_out, int out_size, void* d_ws, size_t ws_size,
                              hipStream_t stream) {
    const float* x      = (const float*)d_in[0];   // (32, 220500)
    const float* wsin   = (const float*)d_in[1];   // (1025, 1, 2048)
    const float* wcos   = (const float*)d_in[2];   // (1025, 1, 2048)
    const float* melb_f = (const float*)d_in[3];   // (128, 1025)
    float* out          = (float*)d_out;           // (32, 128, 431)

    char* ws = (char*)d_ws;
    size_t off = 0;
    auto align256 = [](size_t v) { return (v + 255) & ~(size_t)255; };

    __bf16* xpad = (__bf16*)(ws + off);
    off = align256(off + (size_t)BATCH * XSTRIDE * sizeof(__bf16));
    __bf16* Wc = (__bf16*)(ws + off);
    off = align256(off + (size_t)FPADW * KTOT * sizeof(__bf16));
    __bf16* Ws_ = (__bf16*)(ws + off);
    off = align256(off + (size_t)FPADW * KTOT * sizeof(__bf16));
    __bf16* melb = (__bf16*)(ws + off);
    off = align256(off + (size_t)NMELS * SPADF * sizeof(__bf16));
    (void)ws_size; (void)in_sizes; (void)n_in; (void)out_size;

    {
        int total = BATCH * LPAD;
        k_pad_bf16<<<(total + 255) / 256, 256, 0, stream>>>(x, xpad);
    }
    {
        int total = FPADW * KTOT;
        k_weights_bf16<<<(total + 255) / 256, 256, 0, stream>>>(wsin, wcos, Wc, Ws_);
    }
    {
        int total = NMELS * SPADF;
        k_mel_bf16<<<(total + 255) / 256, 256, 0, stream>>>(melb_f, melb);
    }
    k_melspec<<<BATCH * TILES_PER_B, 256, 0, stream>>>(xpad, Wc, Ws_, melb, out);
}
